// SearchTransfer_87058987089973
// MI455X (gfx1250) — matrix operand order repaired and checked by reading, not compiled, not run on hardware
//
#include <hip/hip_runtime.h>

typedef __attribute__((ext_vector_type(16))) _Float16 v16h;
typedef __attribute__((ext_vector_type(8)))  _Float16 v8h;
typedef __attribute__((ext_vector_type(8)))  float    v8f;

#define B_   4
#define C3_  256
#define H3_  48
#define L_   2304   // 48*48 positions
#define D_   2304   // 256*9 unfolded feature dim
#define KS_  72     // D_/32 WMMA k-steps
#define NT_  144    // L_/16 16-wide tiles

// ---------------- Phase 0: unfold(3,1,1) + L2-normalize rows -> f16 ----------
__global__ void unfold_norm_kernel(const float* __restrict__ img,
                                   const float* __restrict__ ref,
                                   _Float16* __restrict__ Qn,
                                   _Float16* __restrict__ Kn)
{
    __shared__ float row[D_];
    __shared__ float red[256];
    __shared__ float inv_s;

    const int which = blockIdx.y;                 // 0 -> query, 1 -> key
    const float* src = which ? ref : img;
    _Float16* dst    = which ? Kn  : Qn;

    const int bl = blockIdx.x;
    const int b  = bl / L_;
    const int l  = bl % L_;
    const int y  = l / H3_, x = l % H3_;
    const int tid = threadIdx.x;

    float acc = 0.f;
    for (int d = tid; d < D_; d += 256) {
        int c = d / 9, ij = d % 9;
        int i = ij / 3, j = ij % 3;
        int yy = y + i - 1, xx = x + j - 1;
        float v = 0.f;
        if (yy >= 0 && yy < H3_ && xx >= 0 && xx < H3_)
            v = src[((b * C3_ + c) * H3_ + yy) * H3_ + xx];
        row[d] = v;
        acc += v * v;
    }
    red[tid] = acc;
    __syncthreads();
    for (int s = 128; s > 0; s >>= 1) {
        if (tid < s) red[tid] += red[tid + s];
        __syncthreads();
    }
    if (tid == 0) {
        float n = fmaxf(sqrtf(red[0]), 1e-12f);   // F.normalize eps
        inv_s = 1.f / n;
    }
    __syncthreads();
    const float inv = inv_s;
    _Float16* orow = dst + ((size_t)b * L_ + l) * D_;
    for (int d = tid; d < D_; d += 256)
        orow[d] = (_Float16)(row[d] * inv);
}

// ---------------- Phase 1: WMMA GEMM + column max/argmax ----------
// rel[lk][lq] = dot(Kn[lk], Qn[lq]); reduce max over lk per lq.
__global__ void relmax_kernel(const _Float16* __restrict__ Kn,
                              const _Float16* __restrict__ Qn,
                              float* __restrict__ S,       // [B, L] (output 0)
                              int*   __restrict__ relIdx)  // [B, L]
{
    const int b      = blockIdx.y;
    const int lqTile = blockIdx.x;           // 0..143
    const int tid    = threadIdx.x;
    const int wave   = tid >> 5;
    const int lane   = tid & 31;
    const int half   = lane >> 4;            // 0 | 1
    const int l15    = lane & 15;

    const _Float16* Kb = Kn + (size_t)b * L_ * D_;
    const _Float16* Qb = Qn + (size_t)b * L_ * D_;

    // B operand (32x16 KxN): lane holds column n=l15, K chunks at half*8 and half*8+16
    const _Float16* bRow = Qb + (size_t)(lqTile * 16 + l15) * D_ + half * 8;

    float best = -1e30f;
    int   bestIdx = 0;

    for (int tile = wave; tile < NT_; tile += 8) {
        // A operand (16x32 MxK): lane holds row m=l15, K chunks at half*8 and half*8+16
        const _Float16* aRow = Kb + (size_t)(tile * 16 + l15) * D_ + half * 8;
        v8f c = {0.f, 0.f, 0.f, 0.f, 0.f, 0.f, 0.f, 0.f};
        for (int ks = 0; ks < KS_; ++ks) {
            const _Float16* ap = aRow + ks * 32;
            const _Float16* bp = bRow + ks * 32;
            v8h a_lo = *(const v8h*)(ap);        // K = base .. base+7
            v8h a_hi = *(const v8h*)(ap + 16);   // K = base+16 .. base+23
            v8h b_lo = *(const v8h*)(bp);        // K = kb .. kb+7
            v8h b_hi = *(const v8h*)(bp + 16);   // K = kb+16 .. kb+23
            v16h a, bv;
#pragma unroll
            for (int r = 0; r < 8; ++r) {
                a[r] = a_lo[r]; a[8 + r] = a_hi[r];
                bv[r] = b_lo[r]; bv[8 + r] = b_hi[r];
            }
            c = __builtin_amdgcn_wmma_f32_16x16x32_f16(
                    false, a, false, bv, (short)0, c, false, false);
        }
        // C layout: lane n=l15, VGPR r -> M = half*8 + r. Max over this lane's 8 rows.
        float mv = c[0]; int mi = tile * 16 + half * 8;
#pragma unroll
        for (int r = 1; r < 8; ++r) {
            int gi = tile * 16 + half * 8 + r;
            if (c[r] > mv) { mv = c[r]; mi = gi; }
        }
        // merge the other half's 8 rows (same column n)
        float ov = __shfl_xor(mv, 16, 32);
        int   oi = __shfl_xor(mi, 16, 32);
        if (ov > mv || (ov == mv && oi < mi)) { mv = ov; mi = oi; }
        if (mv > best || (mv == best && mi < bestIdx)) { best = mv; bestIdx = mi; }
    }

    __shared__ float sv[8][16];
    __shared__ int   si[8][16];
    if (lane < 16) { sv[wave][l15] = best; si[wave][l15] = bestIdx; }
    __syncthreads();
    if (tid < 16) {
        float v = sv[0][tid]; int i0 = si[0][tid];
        for (int w = 1; w < 8; ++w) {
            float wv = sv[w][tid]; int wi = si[w][tid];
            if (wv > v || (wv == v && wi < i0)) { v = wv; i0 = wi; }
        }
        int lq = lqTile * 16 + tid;
        S[b * L_ + lq]      = v;
        relIdx[b * L_ + lq] = i0;
    }
}

// ---------------- Phase 2: gather-by-argmax + fold-average ----------
// k = 3*scale, pad = scale, stride = scale; out H = 48*scale.
__global__ void transfer_fold_kernel(const float* __restrict__ src,
                                     const int* __restrict__ relIdx,
                                     float* __restrict__ out,
                                     int C, int scale, long long total)
{
    long long idx = (long long)blockIdx.x * blockDim.x + threadIdx.x;
    if (idx >= total) return;
    const int H = H3_ * scale;
    int x = (int)(idx % H); long long t = idx / H;
    int y = (int)(t % H);   t /= H;
    int c = (int)(t % C);   int b = (int)(t / C);

    const int s = scale;
    const int ybase = y / s, yrem = y % s;
    const int xbase = x / s, xrem = x % s;
    float sum = 0.f; int cnt = 0;
#pragma unroll
    for (int ti = 0; ti < 3; ++ti) {
        int ho = ybase + 1 - ti;                  // (y+pad-i)/stride
        if (ho < 0 || ho >= H3_) continue;
        int i = yrem + ti * s;
#pragma unroll
        for (int tj = 0; tj < 3; ++tj) {
            int wo = xbase + 1 - tj;
            if (wo < 0 || wo >= H3_) continue;
            int j = xrem + tj * s;
            ++cnt;
            int m  = relIdx[b * L_ + ho * H3_ + wo];
            int ym = m / H3_, xm = m % H3_;
            int yy = s * ym + i - s;              // zero-padded unfold of ref
            int xx = s * xm + j - s;
            if (yy >= 0 && yy < H && xx >= 0 && xx < H)
                sum += src[(((long long)b * C + c) * H + yy) * H + xx];
        }
    }
    out[idx] = sum / (float)cnt;
}

extern "C" void kernel_launch(void* const* d_in, const int* in_sizes, int n_in,
                              void* d_out, int out_size, void* d_ws, size_t ws_size,
                              hipStream_t stream) {
    const float* dh_img = (const float*)d_in[0];   // [4,256,48,48]
    const float* dh_ref = (const float*)d_in[1];   // [4,256,48,48]
    const float* cl1    = (const float*)d_in[2];   // [4,64,192,192]
    const float* cl2    = (const float*)d_in[3];   // [4,128,96,96]
    const float* cl3    = (const float*)d_in[4];   // [4,256,48,48]
    float* out = (float*)d_out;

    // workspace: Qn (f16) | Kn (f16) | relIdx (int)
    const size_t mat = (size_t)B_ * L_ * D_;
    char* ws = (char*)d_ws;
    _Float16* Qn = (_Float16*)ws;
    _Float16* Kn = Qn + mat;
    int* relIdx  = (int*)(ws + 2 * mat * sizeof(_Float16));

    unfold_norm_kernel<<<dim3(B_ * L_, 2), 256, 0, stream>>>(dh_img, dh_ref, Qn, Kn);

    float* S  = out;                                   // [4,1,48,48]
    float* T3 = S  + (size_t)B_ * L_;                  // [4,256,48,48]
    float* T2 = T3 + (size_t)B_ * 256 * 48 * 48;       // [4,128,96,96]
    float* T1 = T2 + (size_t)B_ * 128 * 96 * 96;       // [4,64,192,192]

    relmax_kernel<<<dim3(NT_, B_), 256, 0, stream>>>(Kn, Qn, S, relIdx);

    {
        long long n = (long long)B_ * 256 * 48 * 48;
        transfer_fold_kernel<<<(unsigned)((n + 255) / 256), 256, 0, stream>>>(cl3, relIdx, T3, 256, 1, n);
    }
    {
        long long n = (long long)B_ * 128 * 96 * 96;
        transfer_fold_kernel<<<(unsigned)((n + 255) / 256), 256, 0, stream>>>(cl2, relIdx, T2, 128, 2, n);
    }
    {
        long long n = (long long)B_ * 64 * 192 * 192;
        transfer_fold_kernel<<<(unsigned)((n + 255) / 256), 256, 0, stream>>>(cl1, relIdx, T1, 64, 4, n);
    }
}
